// TransformerBlock_swiglu_60928406061440
// MI455X (gfx1250) — matrix operand order repaired and checked by reading, not compiled, not run on hardware
//
#include <hip/hip_runtime.h>
#include <math.h>

// Problem constants (match reference).
#define BB 8
#define SS 1024
#define DD 1024
#define HH 16
#define HD 64
#define MM (BB * SS)          // 8192 token rows
#define ROWLEN 1024

typedef __attribute__((ext_vector_type(2))) float v2f;
typedef __attribute__((ext_vector_type(8))) float v8f;
typedef __attribute__((ext_vector_type(8))) __bf16 v8bf;
typedef __attribute__((ext_vector_type(16))) __bf16 v16bf;
typedef __attribute__((ext_vector_type(4))) int v4i;

// --------------------------- CDNA5 async-to-LDS ----------------------------
#if __has_builtin(__builtin_amdgcn_global_load_async_to_lds_b128) && \
    __has_builtin(__builtin_amdgcn_s_wait_asynccnt)
#define HAS_ASYNC_LDS 1
typedef __attribute__((address_space(1))) v4i GV4;  // global int4
typedef __attribute__((address_space(3))) v4i LV4;  // LDS int4
#define WAIT_ASYNC(n) __builtin_amdgcn_s_wait_asynccnt(n)
#else
#define HAS_ASYNC_LDS 0
#define WAIT_ASYNC(n)
#endif

// 16-byte global -> LDS copy. Async (ASYNCcnt-tracked, no VGPR round trip)
// when the toolchain exposes the builtin; synchronous fallback otherwise.
__device__ inline void async_cp16(const void* g, void* l) {
#if HAS_ASYNC_LDS
  __builtin_amdgcn_global_load_async_to_lds_b128((GV4*)g, (LV4*)l, 0, 0);
#else
  *(uint4*)l = *(const uint4*)g;
#endif
}

// fp32 path: D = A(16x4) * B(4x16) + C
__device__ inline v8f wmma_f32(v2f a, v2f b, v8f c) {
  return __builtin_amdgcn_wmma_f32_16x16x4_f32(
      false, a, false, b, (short)0, c, false, false);
}
// bf16 path: D = A(16x32) * B(32x16) + C (fp32 accumulate)
__device__ inline v8f wmma_bf16(v16bf a, v16bf b, v8f c) {
  return __builtin_amdgcn_wmma_f32_16x16x32_bf16(
      false, a, false, b, (short)0, c, false, false);
}

__device__ inline float waveReduceSum(float v) {
#pragma unroll
  for (int off = 16; off > 0; off >>= 1) v += __shfl_xor(v, off);
  return v;
}
__device__ inline float waveReduceMax(float v) {
#pragma unroll
  for (int off = 16; off > 0; off >>= 1) v = fmaxf(v, __shfl_xor(v, off));
  return v;
}

// ---------------------------------------------------------------------------
// Weight convert + transpose: WT[n][k] = bf16(W[k][n]); 64x64 tiles.
// ---------------------------------------------------------------------------
__global__ __launch_bounds__(256) void wtrans_kernel(
    const float* __restrict__ W, __bf16* __restrict__ WT) {
  __shared__ float sh[64][64 + 1];
  const int tid = threadIdx.x;
  const int k0 = blockIdx.y * 64;
  const int n0 = blockIdx.x * 64;
  for (int i = tid; i < 1024; i += 256) {
    const int idx = i * 4;
    const int r = idx >> 6;       // k
    const int c = idx & 63;       // n
    const float4 f = *(const float4*)(W + (size_t)(k0 + r) * DD + n0 + c);
    sh[r][c] = f.x; sh[r][c + 1] = f.y; sh[r][c + 2] = f.z; sh[r][c + 3] = f.w;
  }
  __syncthreads();
  for (int i = tid; i < 512; i += 256) {
    const int idx = i * 8;
    const int n = idx >> 6;
    const int kk = idx & 63;
    union { __bf16 b[8]; uint4 u; } pk;
#pragma unroll
    for (int j = 0; j < 8; ++j) pk.b[j] = (__bf16)sh[kk + j][n];
    *(uint4*)(WT + (size_t)(n0 + n) * DD + k0 + kk) = pk.u;
  }
}

// ---------------------------------------------------------------------------
// LayerNorm (fp32 in -> bf16 out): one block per row of 1024.
// ---------------------------------------------------------------------------
__global__ __launch_bounds__(256) void ln_bf16_kernel(
    const float* __restrict__ x, const float* __restrict__ g,
    const float* __restrict__ b, __bf16* __restrict__ out) {
  const int row = blockIdx.x;
  const float* xr = x + (size_t)row * ROWLEN;
  const int tid = threadIdx.x;
  const int lane = tid & 31;
  const int wave = tid >> 5;
  __shared__ float red[8];

  float vals[4];
  float s = 0.f;
#pragma unroll
  for (int i = 0; i < 4; ++i) {
    vals[i] = xr[tid + i * 256];
    s += vals[i];
  }
  s = waveReduceSum(s);
  if (lane == 0) red[wave] = s;
  __syncthreads();
  float tot = 0.f;
#pragma unroll
  for (int i = 0; i < 8; ++i) tot += red[i];
  const float mean = tot * (1.0f / ROWLEN);
  __syncthreads();

  float vs = 0.f;
#pragma unroll
  for (int i = 0; i < 4; ++i) {
    const float d = vals[i] - mean;
    vs += d * d;
  }
  vs = waveReduceSum(vs);
  if (lane == 0) red[wave] = vs;
  __syncthreads();
  float vtot = 0.f;
#pragma unroll
  for (int i = 0; i < 8; ++i) vtot += red[i];
  const float inv = rsqrtf(vtot * (1.0f / ROWLEN) + 1e-6f);

  __bf16* outr = out + (size_t)row * ROWLEN;
#pragma unroll
  for (int i = 0; i < 4; ++i) {
    const int c = tid + i * 256;
    outr[c] = (__bf16)((vals[i] - mean) * inv * g[c] + b[c]);
  }
}

// ---------------------------------------------------------------------------
// bf16 GEMM: C[M,1024] = A_bf16[M,1024] @ WT_bf16[N,K]^T + bias (+ residual).
// Block tile 64x64, KC=64, 8 waves (4M x 2N), two accumulators per wave.
// Double-buffered LDS staging via GLOBAL_LOAD_ASYNC_TO_LDS_B128: the next
// k-tile streams into LDS (ASYNCcnt) while WMMAs consume the current one.
// OUT_MODE: 0 = fp32 out (Cf), 1 = bf16 out (Cb).
// ---------------------------------------------------------------------------
template <int OUT_MODE, bool HAS_RES>
__global__ __launch_bounds__(256) void gemm_bf16_kernel(
    const __bf16* __restrict__ A, const __bf16* __restrict__ WT,
    const float* __restrict__ bias, const float* __restrict__ res,
    float* __restrict__ Cf, __bf16* __restrict__ Cb) {
  constexpr int KC = 64;
  __shared__ __bf16 As[2][64][KC + 8];  // row stride 144B: aligned+bank-spread
  __shared__ __bf16 Ws[2][64][KC + 8];  // Ws[buf][n][k]

  const int tid = threadIdx.x;
  const int lane = tid & 31;
  const int wave = tid >> 5;
  const int wm = wave & 3;   // 4 waves in M
  const int wn = wave >> 2;  // 2 waves in N
  const int row0 = blockIdx.y * 64;
  const int col0 = blockIdx.x * 64;
  const int m = lane & 15;
  const int koffA = (lane >> 4) * 16;  // A: K = 16*(lane>=16) + 0..15
  const int koffB = (lane >> 4) * 16;  // B: K = 16*(lane>=16) + 0..15

  // Per-thread staging coordinates (2 x 16B per array per tile).
  const int idxA = tid * 8;
  const int r0 = idxA >> 6, c0 = idxA & 63;
  const int idxB = (tid + 256) * 8;
  const int r1 = idxB >> 6, c1 = idxB & 63;

  v8f acc0 = {}, acc1 = {};

  auto prefetch = [&](int buf, int k0) {
    async_cp16(A + (size_t)(row0 + r0) * DD + k0 + c0, &As[buf][r0][c0]);
    async_cp16(A + (size_t)(row0 + r1) * DD + k0 + c1, &As[buf][r1][c1]);
    async_cp16(WT + (size_t)(col0 + r0) * DD + k0 + c0, &Ws[buf][r0][c0]);
    async_cp16(WT + (size_t)(col0 + r1) * DD + k0 + c1, &Ws[buf][r1][c1]);
  };

  constexpr int T = DD / KC;  // 16 k-tiles
  prefetch(0, 0);
  for (int t = 0; t < T; ++t) {
    const int buf = t & 1;
    if (t + 1 < T) {
      prefetch(buf ^ 1, (t + 1) * KC);
      WAIT_ASYNC(4);  // retire current tile's 4 copies; next 4 stay in flight
    } else {
      WAIT_ASYNC(0);
    }
    __syncthreads();
#pragma unroll
    for (int kc = 0; kc < KC; kc += 32) {
      union { v16bf v; v8bf h[2]; } ua, ub0, ub1;
      ua.h[0] = *(const v8bf*)&As[buf][wm * 16 + m][kc + koffA];
      ua.h[1] = *(const v8bf*)&As[buf][wm * 16 + m][kc + koffA + 8];
      ub0.h[0] = *(const v8bf*)&Ws[buf][wn * 32 + m][kc + koffB];
      ub0.h[1] = *(const v8bf*)&Ws[buf][wn * 32 + m][kc + koffB + 8];
      ub1.h[0] = *(const v8bf*)&Ws[buf][wn * 32 + 16 + m][kc + koffB];
      ub1.h[1] = *(const v8bf*)&Ws[buf][wn * 32 + 16 + m][kc + koffB + 8];
      acc0 = wmma_bf16(ua.v, ub0.v, acc0);
      acc1 = wmma_bf16(ua.v, ub1.v, acc1);
    }
    __syncthreads();
  }

  const int crow = row0 + wm * 16 + (lane >> 4) * 8;
  const int ccol = col0 + wn * 32 + (lane & 15);
  const float b0 = bias[ccol];
  const float b1 = bias[ccol + 16];
#pragma unroll
  for (int i = 0; i < 8; ++i) {
    const size_t off = (size_t)(crow + i) * DD + ccol;
    float r0v = acc0[i] + b0;
    float r1v = acc1[i] + b1;
    if (HAS_RES) {
      r0v += res[off];
      r1v += res[off + 16];
    }
    if (OUT_MODE == 0) {
      Cf[off] = r0v;
      Cf[off + 16] = r1v;
    } else {
      Cb[off] = (__bf16)r0v;
      Cb[off + 16] = (__bf16)r1v;
    }
  }
}

// ---------------------------------------------------------------------------
// Scores (fp32 WMMA: precision-critical, feeds exp + validated output):
// for each (b,h): S[q,k] = 0.125 * Q[q,:] . K[k,:]   (head dim 64)
// Tiles staged with async-to-LDS (plain 16B copies).
// ---------------------------------------------------------------------------
__global__ __launch_bounds__(256) void scores_kernel(
    const float* __restrict__ q, const float* __restrict__ k,
    float* __restrict__ wout) {
  const int bh = blockIdx.z;
  const int bb = bh >> 4;
  const int hh = bh & 15;
  const int q0 = blockIdx.y * 32;
  const int k0 = blockIdx.x * 64;

  __shared__ float Qs[32][64 + 4];
  __shared__ float Ks[64][64 + 4];

  const int tid = threadIdx.x;
  for (int i = tid; i < (32 * 64) / 4; i += 256) {
    const int idx = i * 4;
    const int r = idx >> 6;
    const int c = idx & 63;
    async_cp16(q + (size_t)(bb * SS + q0 + r) * DD + hh * HD + c, &Qs[r][c]);
  }
  for (int i = tid; i < (64 * 64) / 4; i += 256) {
    const int idx = i * 4;
    const int r = idx >> 6;
    const int c = idx & 63;
    async_cp16(k + (size_t)(bb * SS + k0 + r) * DD + hh * HD + c, &Ks[r][c]);
  }
  WAIT_ASYNC(0);
  __syncthreads();

  const int lane = tid & 31;
  const int wave = tid >> 5;
  const int wm = wave & 1;
  const int wn = wave >> 1;
  const int mfrag = lane & 15;
  const int kfrag = (lane >> 4) * 2;

  v8f acc = {};
#pragma unroll
  for (int kk = 0; kk < HD; kk += 4) {
    v2f a, b;
    a.x = Qs[wm * 16 + mfrag][kk + kfrag];
    a.y = Qs[wm * 16 + mfrag][kk + kfrag + 1];
    b.x = Ks[wn * 16 + mfrag][kk + kfrag];
    b.y = Ks[wn * 16 + mfrag][kk + kfrag + 1];
    acc = wmma_f32(a, b, acc);
  }

  const int crow = q0 + wm * 16 + (lane >> 4) * 8;
  const int ccol = k0 + wn * 16 + (lane & 15);
#pragma unroll
  for (int i = 0; i < 8; ++i) {
    wout[((size_t)bh * SS + crow + i) * SS + ccol] = acc[i] * 0.125f;
  }
}

// ---------------------------------------------------------------------------
// Softmax in place over rows of length 1024. One block per row.
// ---------------------------------------------------------------------------
__global__ __launch_bounds__(256) void softmax_kernel(float* __restrict__ w) {
  float* row = w + (size_t)blockIdx.x * ROWLEN;
  const int tid = threadIdx.x;
  const int lane = tid & 31;
  const int wave = tid >> 5;
  __shared__ float red[8];

  float v[4];
  float m = -INFINITY;
#pragma unroll
  for (int i = 0; i < 4; ++i) {
    v[i] = row[tid + i * 256];
    m = fmaxf(m, v[i]);
  }
  m = waveReduceMax(m);
  if (lane == 0) red[wave] = m;
  __syncthreads();
  float mt = -INFINITY;
#pragma unroll
  for (int i = 0; i < 8; ++i) mt = fmaxf(mt, red[i]);
  __syncthreads();

  float s = 0.f;
#pragma unroll
  for (int i = 0; i < 4; ++i) {
    v[i] = __expf(v[i] - mt);
    s += v[i];
  }
  s = waveReduceSum(s);
  if (lane == 0) red[wave] = s;
  __syncthreads();
  float st = 0.f;
#pragma unroll
  for (int i = 0; i < 8; ++i) st += red[i];
  const float inv = 1.0f / st;
#pragma unroll
  for (int i = 0; i < 4; ++i) row[tid + i * 256] = v[i] * inv;
}

// ---------------------------------------------------------------------------
// PV (bf16 WMMA): attn[q,d] = sum_key W[q,key] * V[key,d] per (b,h).
// Block: 64 q x 64 d (full head), KC = 64 keys; weights converted to bf16
// during staging; V tile transposed in LDS to [d][key]. (Staging keeps the
// synchronous path: it performs conversion/transpose, not a straight copy.)
// ---------------------------------------------------------------------------
__global__ __launch_bounds__(256) void pv_bf16_kernel(
    const float* __restrict__ w, const __bf16* __restrict__ v,
    __bf16* __restrict__ attn) {
  const int bh = blockIdx.z;
  const int bb = bh >> 4;
  const int hh = bh & 15;
  const int q0 = blockIdx.y * 64;

  __shared__ __bf16 As[64][64 + 8];  // [q][key]
  __shared__ __bf16 Vs[64][64 + 8];  // [d][key]

  const int tid = threadIdx.x;
  const int lane = tid & 31;
  const int wave = tid >> 5;
  const int wm = wave & 3;
  const int wn = wave >> 2;
  const int m = lane & 15;
  const int koffA = (lane >> 4) * 16;
  const int koffB = (lane >> 4) * 16;

  v8f acc0 = {}, acc1 = {};
  for (int k0 = 0; k0 < SS; k0 += 64) {
    for (int i = tid; i < 1024; i += 256) {  // weights 64x64 f32 -> bf16
      const int idx = i * 4;
      const int r = idx >> 6;
      const int c = idx & 63;
      const float4 f =
          *(const float4*)(w + ((size_t)bh * SS + q0 + r) * SS + k0 + c);
      As[r][c] = (__bf16)f.x;
      As[r][c + 1] = (__bf16)f.y;
      As[r][c + 2] = (__bf16)f.z;
      As[r][c + 3] = (__bf16)f.w;
    }
    for (int i = tid; i < 512; i += 256) {  // V 64key x 64d -> Vs[d][key]
      const int idx = i * 8;
      const int r = idx >> 6;  // key
      const int c = idx & 63;  // d base
      const v8bf vv =
          *(const v8bf*)(v + (size_t)(bb * SS + k0 + r) * DD + hh * HD + c);
#pragma unroll
      for (int j = 0; j < 8; ++j) Vs[c + j][r] = vv[j];
    }
    __syncthreads();
#pragma unroll
    for (int kc = 0; kc < 64; kc += 32) {
      union { v16bf v; v8bf h[2]; } ua, ub0, ub1;
      ua.h[0] = *(const v8bf*)&As[wm * 16 + m][kc + koffA];
      ua.h[1] = *(const v8bf*)&As[wm * 16 + m][kc + koffA + 8];
      ub0.h[0] = *(const v8bf*)&Vs[wn * 32 + m][kc + koffB];
      ub0.h[1] = *(const v8bf*)&Vs[wn * 32 + m][kc + koffB + 8];
      ub1.h[0] = *(const v8bf*)&Vs[wn * 32 + 16 + m][kc + koffB];
      ub1.h[1] = *(const v8bf*)&Vs[wn * 32 + 16 + m][kc + koffB + 8];
      acc0 = wmma_bf16(ua.v, ub0.v, acc0);
      acc1 = wmma_bf16(ua.v, ub1.v, acc1);
    }
    __syncthreads();
  }

  const int crow = q0 + wm * 16 + (lane >> 4) * 8;
  const int ccol = wn * 32 + (lane & 15);
#pragma unroll
  for (int i = 0; i < 8; ++i) {
    const size_t off = (size_t)(bb * SS + crow + i) * DD + hh * HD;
    attn[off + ccol] = (__bf16)acc0[i];
    attn[off + ccol + 16] = (__bf16)acc1[i];
  }
}

// ---------------------------------------------------------------------------
// SwiGLU gate: g = vy * wy * sigmoid(beta * wy)   (fp32 in -> bf16 out)
// ---------------------------------------------------------------------------
__global__ __launch_bounds__(256) void gate_kernel(
    const float* __restrict__ wy, const float* __restrict__ vy,
    const float* __restrict__ beta, __bf16* __restrict__ g, int n) {
  const int i = blockIdx.x * 256 + threadIdx.x;
  if (i < n) {
    const float w = wy[i];
    const float s = 1.0f / (1.0f + __expf(-beta[0] * w));
    g[i] = (__bf16)(vy[i] * (w * s));
  }
}

// ---------------------------------------------------------------------------
extern "C" void kernel_launch(void* const* d_in, const int* in_sizes, int n_in,
                              void* d_out, int out_size, void* d_ws,
                              size_t ws_size, hipStream_t stream) {
  (void)in_sizes; (void)n_in; (void)out_size; (void)ws_size;

  const float* x     = (const float*)d_in[0];
  const float* ln1_g = (const float*)d_in[1];
  const float* ln1_b = (const float*)d_in[2];
  const float* wq    = (const float*)d_in[3];
  const float* bq    = (const float*)d_in[4];
  const float* wk    = (const float*)d_in[5];
  const float* bk    = (const float*)d_in[6];
  const float* wv    = (const float*)d_in[7];
  const float* bv_   = (const float*)d_in[8];
  const float* wo    = (const float*)d_in[9];
  const float* bo    = (const float*)d_in[10];
  const float* ln2_g = (const float*)d_in[11];
  const float* ln2_b = (const float*)d_in[12];
  const float* sw_W  = (const float*)d_in[13];
  const float* sw_Wb = (const float*)d_in[14];
  const float* sw_V  = (const float*)d_in[15];
  const float* sw_Vb = (const float*)d_in[16];
  const float* beta  = (const float*)d_in[17];
  const float* d1_w  = (const float*)d_in[18];
  const float* d1_b  = (const float*)d_in[19];

  float* out = (float*)d_out;
  const size_t SL = (size_t)BB * SS * DD;  // 8M elements
  float* wts = out + SL;                   // [B,H,S,S] weights output region

  // Workspace layout (bytes).
  char* w8 = (char*)d_ws;
  const size_t MB = 1u << 20;
  __bf16* wqT  = (__bf16*)(w8 + 0 * MB);
  __bf16* wkT  = (__bf16*)(w8 + 2 * MB);
  __bf16* wvT  = (__bf16*)(w8 + 4 * MB);
  __bf16* woT  = (__bf16*)(w8 + 6 * MB);
  __bf16* swWT = (__bf16*)(w8 + 8 * MB);
  __bf16* swVT = (__bf16*)(w8 + 10 * MB);
  __bf16* d1T  = (__bf16*)(w8 + 12 * MB);
  __bf16* actA = (__bf16*)(w8 + 14 * MB);  // h -> attn -> y -> gate (16MB)
  __bf16* v_bf = (__bf16*)(w8 + 30 * MB);  // v projection, bf16 (16MB)
  float*  qf   = (float*)(w8 + 46 * MB);   // q -> wy (32MB)
  float*  kf   = (float*)(w8 + 78 * MB);   // k -> vy (32MB)
  float*  xres = (float*)(w8 + 110 * MB);  // residual (32MB)

  dim3 blk(256);
  dim3 gWT(16, 16);                         // weight transpose tiles
  dim3 gG(DD / 64, MM / 64);                // 16 x 128 GEMM blocks
  dim3 gScores(SS / 64, SS / 32, BB * HH);  // 16 x 32 x 128
  dim3 gPV(1, SS / 64, BB * HH);            // 1 x 16 x 128

  // One-time weight convert+transpose (bf16 [N][K]).
  wtrans_kernel<<<gWT, blk, 0, stream>>>(wq, wqT);
  wtrans_kernel<<<gWT, blk, 0, stream>>>(wk, wkT);
  wtrans_kernel<<<gWT, blk, 0, stream>>>(wv, wvT);
  wtrans_kernel<<<gWT, blk, 0, stream>>>(wo, woT);
  wtrans_kernel<<<gWT, blk, 0, stream>>>(sw_W, swWT);
  wtrans_kernel<<<gWT, blk, 0, stream>>>(sw_V, swVT);
  wtrans_kernel<<<gWT, blk, 0, stream>>>(d1_w, d1T);

  // h = LN1(x)  (bf16)
  ln_bf16_kernel<<<MM, blk, 0, stream>>>(x, ln1_g, ln1_b, actA);
  // q,k fp32 (feed fp32 scores); v bf16 (feeds bf16 PV)
  gemm_bf16_kernel<0, false><<<gG, blk, 0, stream>>>(actA, wqT, bq, nullptr, qf, nullptr);
  gemm_bf16_kernel<0, false><<<gG, blk, 0, stream>>>(actA, wkT, bk, nullptr, kf, nullptr);
  gemm_bf16_kernel<1, false><<<gG, blk, 0, stream>>>(actA, wvT, bv_, nullptr, nullptr, v_bf);
  // scores (fp32 WMMA) -> softmax in place in the weights output region
  scores_kernel<<<gScores, blk, 0, stream>>>(qf, kf, wts);
  softmax_kernel<<<BB * HH * SS, blk, 0, stream>>>(wts);
  // attn = weights @ V  (bf16 WMMA, bf16 out into actA)
  pv_bf16_kernel<<<gPV, blk, 0, stream>>>(wts, v_bf, actA);
  // xres = x + attn @ wo + bo  (fp32 out)
  gemm_bf16_kernel<0, true><<<gG, blk, 0, stream>>>(actA, woT, bo, x, xres, nullptr);
  // y = LN2(xres)  (bf16)
  ln_bf16_kernel<<<MM, blk, 0, stream>>>(xres, ln2_g, ln2_b, actA);
  // wy, vy (fp32: gate consumes fp32)
  gemm_bf16_kernel<0, false><<<gG, blk, 0, stream>>>(actA, swWT, sw_Wb, nullptr, qf, nullptr);
  gemm_bf16_kernel<0, false><<<gG, blk, 0, stream>>>(actA, swVT, sw_Vb, nullptr, kf, nullptr);
  // g = vy * swish(wy)  (bf16)
  gate_kernel<<<(int)((SL + 255) / 256), blk, 0, stream>>>(qf, kf, beta, actA, (int)SL);
  // out = xres + g @ d1_w + d1_b
  gemm_bf16_kernel<0, true><<<gG, blk, 0, stream>>>(actA, d1T, d1_b, xres, out, nullptr);
}
